// PathAttention_25400436588751
// MI455X (gfx1250) — matrix operand order repaired and checked by reading, not compiled, not run on hardware
//
#include <hip/hip_runtime.h>
#include <stdint.h>

typedef __bf16 bf16_t;
typedef __attribute__((ext_vector_type(16))) __bf16 v16bf;
typedef __attribute__((ext_vector_type(8)))  __bf16 v8bf;
typedef __attribute__((ext_vector_type(8)))  float  v8f;

#define K_DMODEL 512
#define K_HEADS  8
#define K_HD     64
#define K_B      4
#define K_L      1024
#define K_S      1024
#define K_BH     32
#define OUT_ELEMS 2097152          /* B*L*D_MODEL */
#define LDSP     40                /* padded LDS row stride in bf16 (80B, 16B-aligned) */

/* ---------------- CDNA5 async global->LDS copy (ASYNCcnt) -------------------- */
/* LDS byte offset == low 32 bits of generic __shared__ pointer (flat-LDS        */
/* aperture).                                             */
__device__ __forceinline__ void async_b128(void* lds, const void* g) {
  asm volatile("global_load_async_to_lds_b128 %0, %1, off"
               :: "v"((uint32_t)(uintptr_t)lds),
                  "v"((uint64_t)(uintptr_t)g)
               : "memory");
}
__device__ __forceinline__ void async_wait() {
  asm volatile("s_wait_asynccnt 0x0" ::: "memory");
}

/* ---------------- WMMA fragment loads (CDNA5 ISA 7.12.2 layouts) -------------- */
__device__ __forceinline__ v16bf frag_a(const bf16_t* row, int half) {
  v8bf a0 = *(const v8bf*)(row + half * 8);
  v8bf a1 = *(const v8bf*)(row + 16 + half * 8);
  v16bf r;
#pragma unroll
  for (int e = 0; e < 8; ++e) { r[e] = a0[e]; r[e + 8] = a1[e]; }
  return r;
}
__device__ __forceinline__ v16bf frag_b(const bf16_t* row, int half) {
  v8bf b0 = *(const v8bf*)(row + half * 8);
  v8bf b1 = *(const v8bf*)(row + 16 + half * 8);
  v16bf r;
#pragma unroll
  for (int e = 0; e < 8; ++e) { r[e] = b0[e]; r[e + 8] = b1[e]; }
  return r;
}

/* 8 waves * (16x64) = 128x64 block tile, K-step 32, 4 wmma per wave per step */
__device__ __forceinline__ void wmma_step(const bf16_t (*As)[LDSP], const bf16_t (*Bt)[LDSP],
                                          int w, int lane, v8f acc[4]) {
  int half = lane >> 4, l = lane & 15;
  v16bf a = frag_a(As[w * 16 + l], half);
#pragma unroll
  for (int tn = 0; tn < 4; ++tn) {
    v16bf b = frag_b(Bt[tn * 16 + l], half);
    acc[tn] = __builtin_amdgcn_wmma_f32_16x16x32_bf16(false, a, false, b, (short)0,
                                                      acc[tn], false, false);
  }
}

/* ---------------- staging helpers ------------------------------------------- */
/* As (128x32 bf16) from contiguous bf16 rows: 512 x 16B chunks, 2 per thread */
__device__ __forceinline__ void stage_a_async(bf16_t (*As)[LDSP], const bf16_t* src,
                                              int lda, int t) {
#pragma unroll
  for (int i = 0; i < 2; ++i) {
    int cid = t + (i << 8);
    int r = cid >> 2, co = (cid & 3) << 3;
    async_b128(&As[r][co], src + (size_t)r * lda + co);
  }
}
/* Bt (64x32 bf16) from contiguous bf16 rows: 256 x 16B chunks, 1 per thread */
__device__ __forceinline__ void stage_bt_async(bf16_t (*Bt)[LDSP], const bf16_t* src,
                                               int ldb, int t) {
  int n = t >> 2, co = (t & 3) << 3;
  async_b128(&Bt[n][co], src + (size_t)n * ldb + co);
}
/* As (128x32) from fp32 rows with convert: 16 floats per thread, vectorized */
__device__ __forceinline__ void stage_a_cvt(bf16_t (*As)[LDSP], const float* src,
                                            int lda, int t) {
  int r = t >> 1, c0 = (t & 1) << 4;
  const float4* s = (const float4*)(src + (size_t)r * lda + c0);
  float4 f0 = s[0], f1 = s[1], f2 = s[2], f3 = s[3];
  v8bf p0, p1;
  p0[0]=(bf16_t)f0.x; p0[1]=(bf16_t)f0.y; p0[2]=(bf16_t)f0.z; p0[3]=(bf16_t)f0.w;
  p0[4]=(bf16_t)f1.x; p0[5]=(bf16_t)f1.y; p0[6]=(bf16_t)f1.z; p0[7]=(bf16_t)f1.w;
  p1[0]=(bf16_t)f2.x; p1[1]=(bf16_t)f2.y; p1[2]=(bf16_t)f2.z; p1[3]=(bf16_t)f2.w;
  p1[4]=(bf16_t)f3.x; p1[5]=(bf16_t)f3.y; p1[6]=(bf16_t)f3.z; p1[7]=(bf16_t)f3.w;
  *(v8bf*)&As[r][c0]     = p0;
  *(v8bf*)&As[r][c0 + 8] = p1;
}

/* ===== Kernel 0: weights -> bf16, transposed to [n][k] (one-time, 2MB) ======= */
__global__ __launch_bounds__(256) void k_wtr(
    const float* __restrict__ Wq, const float* __restrict__ Wk,
    const float* __restrict__ Wv, const float* __restrict__ Wo,
    bf16_t* __restrict__ Wtq, bf16_t* __restrict__ Wtk,
    bf16_t* __restrict__ Wtv, bf16_t* __restrict__ Wto) {
  int which = blockIdx.y;
  const float* W = which == 0 ? Wq : (which == 1 ? Wk : (which == 2 ? Wv : Wo));
  bf16_t* Wt = which == 0 ? Wtq : (which == 1 ? Wtk : (which == 2 ? Wtv : Wto));
  int gid = blockIdx.x * 256 + threadIdx.x;   /* 32768 threads: 8 elems each */
  int k = gid >> 6, n0 = (gid & 63) << 3;
  const float4* s = (const float4*)(W + (size_t)k * K_DMODEL + n0);
  float4 a = s[0], b = s[1];
  float v[8] = {a.x, a.y, a.z, a.w, b.x, b.y, b.z, b.w};
#pragma unroll
  for (int e = 0; e < 8; ++e)
    Wt[(size_t)(n0 + e) * K_DMODEL + k] = (bf16_t)v[e];
}

/* ===== Kernel 1: fused QKV projection + head split (v stored transposed) ===== */
__global__ __launch_bounds__(256) void k_qkv_proj(
    const float* __restrict__ queries, const float* __restrict__ keys,
    const float* __restrict__ values,
    const bf16_t* __restrict__ Wtq, const bf16_t* __restrict__ Wtk,
    const bf16_t* __restrict__ Wtv,
    bf16_t* __restrict__ qh, bf16_t* __restrict__ kh, bf16_t* __restrict__ vt) {
  __shared__ alignas(16) bf16_t As[128][LDSP];
  __shared__ alignas(16) bf16_t Bt[64][LDSP];
  int which = blockIdx.z;
  const float* X = which == 0 ? queries : (which == 1 ? keys : values);
  const bf16_t* Wt = which == 0 ? Wtq : (which == 1 ? Wtk : Wtv);
  bf16_t* dstqk = which == 0 ? qh : kh;
  int m0 = blockIdx.x * 128, n0 = blockIdx.y * 64;
  int t = threadIdx.x, w = t >> 5, lane = t & 31, half = lane >> 4, l16 = lane & 15;
  v8f acc[4] = {};
  for (int k0 = 0; k0 < K_DMODEL; k0 += 32) {
    stage_bt_async(Bt, Wt + (size_t)n0 * K_DMODEL + k0, K_DMODEL, t);
    stage_a_cvt(As, X + (size_t)m0 * K_DMODEL + k0, K_DMODEL, t);
    if (k0 + 32 < K_DMODEL)
      __builtin_prefetch(X + (size_t)(m0 + (t >> 1)) * K_DMODEL + k0 + 32, 0, 1);
    async_wait();
    __syncthreads();
    wmma_step(As, Bt, w, lane, acc);
    __syncthreads();
  }
#pragma unroll
  for (int tn = 0; tn < 4; ++tn)
#pragma unroll
    for (int r = 0; r < 8; ++r) {
      int m = m0 + w * 16 + half * 8 + r;     /* C: M = r + half*8, N = lane&15 */
      int n = n0 + tn * 16 + l16;
      int b = m >> 10, l = m & 1023, h = n >> 6, d = n & 63;
      bf16_t val = (bf16_t)acc[tn][r];
      size_t bh = (size_t)(b << 3) + h;
      if (which == 2)
        vt[(bh << 16) + ((size_t)d << 10) + l] = val;          /* [bh][d][s]  */
      else
        dstqk[((bh << 10) + l) * K_HD + d] = val;              /* [bh][l][d]  */
    }
}

/* ===== Kernel 2: scores = q·kᵀ / 8 -> fp32 in attn region (all-async stage) == */
__global__ __launch_bounds__(256) void k_scores(
    const bf16_t* __restrict__ qh, const bf16_t* __restrict__ kh,
    float* __restrict__ scores) {
  __shared__ alignas(16) bf16_t As[128][LDSP];
  __shared__ alignas(16) bf16_t Bt[64][LDSP];
  int bh = blockIdx.z;
  const bf16_t* Q = qh + ((size_t)bh << 16);
  const bf16_t* K = kh + ((size_t)bh << 16);
  float* out = scores + ((size_t)bh << 20);
  int m0 = blockIdx.x * 128, n0 = blockIdx.y * 64;
  int t = threadIdx.x, w = t >> 5, lane = t & 31, half = lane >> 4, l16 = lane & 15;
  v8f acc[4] = {};
  for (int k0 = 0; k0 < K_HD; k0 += 32) {
    stage_a_async(As, Q + (size_t)m0 * K_HD + k0, K_HD, t);
    stage_bt_async(Bt, K + (size_t)n0 * K_HD + k0, K_HD, t);   /* Bt[s][d]    */
    async_wait();
    __syncthreads();
    wmma_step(As, Bt, w, lane, acc);
    __syncthreads();
  }
#pragma unroll
  for (int tn = 0; tn < 4; ++tn)
#pragma unroll
    for (int r = 0; r < 8; ++r) {
      int m = m0 + w * 16 + half * 8 + r;
      int n = n0 + tn * 16 + l16;
      out[((size_t)m << 10) + n] = acc[tn][r] * 0.125f;        /* 1/sqrt(64)  */
    }
}

/* ==== Kernel 3: path_mean[i,j] = mean_{b,h} S[i,m1]*S[m2,j]  (L2-resident) === */
__global__ __launch_bounds__(256) void k_pathmean(const float* __restrict__ scores,
                                                  float* __restrict__ pm) {
  int idx = blockIdx.x * 256 + threadIdx.x;
  int i = idx >> 10, j = idx & 1023;
  int d = j - i;
  int fd1 = (d >= 0) ? (d / 3) : -((-d + 2) / 3);              /* floor(d/3)  */
  int t2 = 2 * d;
  int fd2 = (t2 >= 0) ? (t2 / 3) : -((-t2 + 2) / 3);           /* floor(2d/3) */
  int m1 = i + fd1, m2 = i + fd2;
  float s = 0.f;
#pragma unroll 4
  for (int bh = 0; bh < K_BH; ++bh) {
    const float* plane = scores + ((size_t)bh << 20);
    s += plane[((size_t)i << 10) + m1] * plane[((size_t)m2 << 10) + j];
  }
  pm[idx] = s * (1.0f / 32.0f);
}

/* ==== Kernel 4: in-place mask(near)/path_mean substitution + row softmax ===== */
__global__ __launch_bounds__(256) void k_softmax(float* __restrict__ attn,
                                                 const float* __restrict__ pm) {
  int row = blockIdx.x;                  /* row = bh*1024 + i */
  int i = row & 1023;
  float* srow = attn + ((size_t)row << 10);
  const float* pmrow = pm + ((size_t)i << 10);
  __shared__ float red[256];
  int t = threadIdx.x;
  float vals[4], mx = -3.402823466e+38f;
#pragma unroll
  for (int k = 0; k < 4; ++k) {
    int j = t + (k << 8);
    int ad = j - i; ad = ad < 0 ? -ad : ad;
    float v = (ad <= 3) ? srow[j] : pmrow[j];
    vals[k] = v; mx = fmaxf(mx, v);
  }
  red[t] = mx; __syncthreads();
  for (int off = 128; off > 0; off >>= 1) {
    if (t < off) red[t] = fmaxf(red[t], red[t + off]);
    __syncthreads();
  }
  mx = red[0]; __syncthreads();
  float sum = 0.f;
#pragma unroll
  for (int k = 0; k < 4; ++k) { vals[k] = __expf(vals[k] - mx); sum += vals[k]; }
  red[t] = sum; __syncthreads();
  for (int off = 128; off > 0; off >>= 1) {
    if (t < off) red[t] += red[t + off];
    __syncthreads();
  }
  float inv = 1.0f / red[0];
#pragma unroll
  for (int k = 0; k < 4; ++k) srow[t + (k << 8)] = vals[k] * inv;
}

/* ===== Kernel 5: out_h = attn @ v (v pre-transposed), concat heads (bf16) ==== */
__global__ __launch_bounds__(256) void k_av(const float* __restrict__ attn,
                                            const bf16_t* __restrict__ vt,
                                            bf16_t* __restrict__ oh) {
  __shared__ alignas(16) bf16_t As[128][LDSP];
  __shared__ alignas(16) bf16_t Bt[64][LDSP];
  int bh = blockIdx.z;
  const float* A = attn + ((size_t)bh << 20);
  const bf16_t* V = vt + ((size_t)bh << 16);     /* [d][s] rows               */
  int m0 = blockIdx.x * 128;
  int t = threadIdx.x, w = t >> 5, lane = t & 31, half = lane >> 4, l16 = lane & 15;
  v8f acc[4] = {};
  for (int k0 = 0; k0 < K_S; k0 += 32) {
    stage_bt_async(Bt, V + k0, K_S, t);          /* Bt[d][s] contiguous       */
    stage_a_cvt(As, A + (size_t)m0 * K_S + k0, K_S, t);
    if (k0 + 32 < K_S)
      __builtin_prefetch(A + (size_t)(m0 + (t >> 1)) * K_S + k0 + 32, 0, 1);
    async_wait();
    __syncthreads();
    wmma_step(As, Bt, w, lane, acc);
    __syncthreads();
  }
  int b = bh >> 3, h = bh & 7;
#pragma unroll
  for (int tn = 0; tn < 4; ++tn)
#pragma unroll
    for (int r = 0; r < 8; ++r) {
      int l = m0 + w * 16 + half * 8 + r;
      int d = tn * 16 + l16;
      oh[(((size_t)b << 10) + l) * K_DMODEL + h * K_HD + d] = (bf16_t)acc[tn][r];
    }
}

/* ===== Kernel 6: out = oh @ W_o  (fp32 result, all-async staging) ============ */
__global__ __launch_bounds__(256) void k_outproj(const bf16_t* __restrict__ oh,
                                                 const bf16_t* __restrict__ Wto,
                                                 float* __restrict__ out) {
  __shared__ alignas(16) bf16_t As[128][LDSP];
  __shared__ alignas(16) bf16_t Bt[64][LDSP];
  int m0 = blockIdx.x * 128, n0 = blockIdx.y * 64;
  int t = threadIdx.x, w = t >> 5, lane = t & 31, half = lane >> 4, l16 = lane & 15;
  v8f acc[4] = {};
  for (int k0 = 0; k0 < K_DMODEL; k0 += 32) {
    stage_a_async(As, oh + (size_t)m0 * K_DMODEL + k0, K_DMODEL, t);
    stage_bt_async(Bt, Wto + (size_t)n0 * K_DMODEL + k0, K_DMODEL, t);
    async_wait();
    __syncthreads();
    wmma_step(As, Bt, w, lane, acc);
    __syncthreads();
  }
#pragma unroll
  for (int tn = 0; tn < 4; ++tn)
#pragma unroll
    for (int r = 0; r < 8; ++r) {
      int m = m0 + w * 16 + half * 8 + r;
      int n = n0 + tn * 16 + l16;
      out[(size_t)m * K_DMODEL + n] = acc[tn][r];
    }
}

extern "C" void kernel_launch(void* const* d_in, const int* in_sizes, int n_in,
                              void* d_out, int out_size, void* d_ws, size_t ws_size,
                              hipStream_t stream) {
  (void)in_sizes; (void)n_in; (void)out_size; (void)ws_size;
  const float* queries = (const float*)d_in[0];
  const float* keys    = (const float*)d_in[1];
  const float* values  = (const float*)d_in[2];
  const float* Wq      = (const float*)d_in[3];
  const float* Wk      = (const float*)d_in[4];
  const float* Wv      = (const float*)d_in[5];
  const float* Wo      = (const float*)d_in[6];

  char* ws = (char*)d_ws;
  bf16_t* qh  = (bf16_t*)(ws + 0);
  bf16_t* kh  = (bf16_t*)(ws + (4u  << 20));
  bf16_t* vt  = (bf16_t*)(ws + (8u  << 20));   /* v transposed: [bh][d][s]    */
  float*  pm  = (float*) (ws + (12u << 20));
  bf16_t* oh  = (bf16_t*)(ws + (16u << 20));
  bf16_t* Wtq = (bf16_t*)(ws + (20u << 20));   /* bf16 weights, [n][k]        */
  bf16_t* Wtk = (bf16_t*)(ws + (20u << 20) + (512u << 10));
  bf16_t* Wtv = (bf16_t*)(ws + (21u << 20));
  bf16_t* Wto = (bf16_t*)(ws + (21u << 20) + (512u << 10));

  float* out  = (float*)d_out;
  float* attn = out + OUT_ELEMS;   /* scores built in place, then softmaxed    */

  dim3 blk(256);
  /* 0) bf16-transpose all weights (one-time) */
  k_wtr<<<dim3(128, 4, 1), blk, 0, stream>>>(Wq, Wk, Wv, Wo, Wtq, Wtk, Wtv, Wto);
  /* 1) q/k/v projections: (4096x512)@(512x512), z selects q/k/v; v transposed */
  k_qkv_proj<<<dim3(32, 8, 3), blk, 0, stream>>>(queries, keys, values,
                                                 Wtq, Wtk, Wtv, qh, kh, vt);
  /* 2) scores per (b,h): (1024x64)@(64x1024)/8 -> attn region of d_out */
  k_scores<<<dim3(8, 16, K_BH), blk, 0, stream>>>(qh, kh, attn);
  /* 3) path_mean over 32 (b,h) planes (whole score tensor is L2-resident) */
  k_pathmean<<<dim3((K_L * K_S) / 256), blk, 0, stream>>>(attn, pm);
  /* 4) near-mask / path_mean substitution + softmax, in place */
  k_softmax<<<dim3(K_BH * K_L), blk, 0, stream>>>(attn, pm);
  /* 5) attn @ v -> head-concat bf16 activations */
  k_av<<<dim3(8, 1, K_BH), blk, 0, stream>>>(attn, vt, oh);
  /* 6) final projection with W_o -> fp32 out */
  k_outproj<<<dim3(32, 8, 1), blk, 0, stream>>>(oh, Wto, out);
}
